// HFEN3D_31593779429667
// MI455X (gfx1250) — matrix operand order repaired and checked by reading, not compiled, not run on hardware
//
#include <hip/hip_runtime.h>
#include <hip/hip_bf16.h>
#include <math.h>

typedef __attribute__((ext_vector_type(16))) _Float16 v16h;
typedef __attribute__((ext_vector_type(8)))  float    v8f;

#define KS 15
#define PAD 7
#define ZT 4                 // z-outputs per block
#define NSLAB (ZT + KS - 1)  // 18 staged z-planes
#define DD 160
#define HH 160
#define WW 160
#define NB 4
// blocks: NB * (DD/ZT) * (HH/16) * (WW/16) = 4*40*10*10
#define NBLOCKS 16000
#define BTAB_ELEMS (KS*KS*32*16)   // 115200 f16 = 230400 bytes

// ---------------------------------------------------------------------------
// Phase 0: LoG weights + pre-expanded Toeplitz B fragments in WMMA B layout.
// B (32x16) per (dz,dy): lane<16 holds K=0..15 of column n=lane; lane>=16
// holds K=16..31 of column n=lane-16; element e -> K = kbase+e.
// ---------------------------------------------------------------------------
__global__ __launch_bounds__(256) void hfen_build_btable(_Float16* __restrict__ btab) {
    __shared__ float w[KS*KS*KS];
    const float SIG = 1.5f;
    if (threadIdx.x == 0) {
        float s = 0.f;
        for (int i = 0; i < KS; ++i)
            for (int j = 0; j < KS; ++j)
                for (int k = 0; k < KS; ++k) {
                    float x = (float)(i - PAD), y = (float)(j - PAD), z = (float)(k - PAD);
                    float rr2 = x*x + y*y + z*z;
                    float g = expf(-rr2 / (2.f * SIG * SIG));
                    w[(i*KS + j)*KS + k] = g;
                    s += g;
                }
        float inv = 1.f / s;
        float mean = 0.f;
        for (int idx = 0; idx < KS*KS*KS; ++idx) {
            int i = idx / (KS*KS), j = (idx / KS) % KS, k = idx % KS;
            float x = (float)(i - PAD), y = (float)(j - PAD), z = (float)(k - PAD);
            float rr2 = x*x + y*y + z*z;
            float v = (rr2 - 3.f*SIG*SIG) / (SIG*SIG*SIG*SIG) * (w[idx] * inv);
            w[idx] = v;
            mean += v;
        }
        mean *= (1.f / (float)(KS*KS*KS));
        for (int idx = 0; idx < KS*KS*KS; ++idx) w[idx] -= mean;
    }
    __syncthreads();
    for (int p = threadIdx.x; p < KS*KS; p += blockDim.x) {   // p = dz*KS+dy
        const float* w15 = &w[p * KS];
        for (int lane = 0; lane < 32; ++lane) {
            int n     = lane & 15;
            int kbase = (lane >> 4) * 16;
            for (int e = 0; e < 16; ++e) {
                int d = (kbase + e) - n;
                float v = (d >= 0 && d < KS) ? w15[d] : 0.f;
                btab[((size_t)p * 32 + lane) * 16 + e] = (_Float16)v;
            }
        }
    }
}

// ---------------------------------------------------------------------------
// Phase 1: one wave per 16x16 (y,x) tile x ZT z-outputs; two passes (pred,
// target) reusing LDS. Inner loops are branch-free -> in-place WMMA chains.
// Slab columns stored in natural order (K = 16*half + e) so each lane's
// 16-element A fragment is one contiguous 32B v16h LDS load.
// ---------------------------------------------------------------------------
__global__ __launch_bounds__(32) void hfen_conv(
    const float* __restrict__ pred, const float* __restrict__ target,
    const _Float16* __restrict__ btab, float* __restrict__ partial) {

    // NSLAB * 30 rows * 2 chunks of 16 f16 = 34,560 bytes
    __shared__ v16h slabv[NSLAB * 30 * 2];
    _Float16* slabh = (_Float16*)slabv;

    int bid = blockIdx.x;
    int xt = bid % 10; bid /= 10;
    int yt = bid % 10; bid /= 10;
    int zt = bid % (DD / ZT); bid /= (DD / ZT);
    int b  = bid;
    const int z0 = zt * ZT, y0 = yt * 16, x0 = xt * 16;

    const int lane = threadIdx.x;
    const int m    = lane & 15;
    const int half = lane >> 4;
    // identity column mapping for staging (lane == source K index == column)
    const int lcol = lane;
    const int x    = x0 - PAD + lane;
    const bool xok = (unsigned)x < (unsigned)WW;

    const float* bases[2] = { pred   + (size_t)b * DD * HH * WW,
                              target + (size_t)b * DD * HH * WW };

    v8f acc[2][ZT];
#pragma unroll
    for (int t = 0; t < 2; ++t)
#pragma unroll
        for (int zo = 0; zo < ZT; ++zo) acc[t][zo] = {};

    for (int t = 0; t < 2; ++t) {
        // ---- stage all NSLAB z-planes (30 x 32 f16, identity col order) ----
        const float* base = bases[t];
        for (int s = 0; s < NSLAB; ++s) {
            int zsrc = z0 - PAD + s;
            bool zok = (unsigned)zsrc < (unsigned)DD;
            const float* src = base + (size_t)(zok ? zsrc : 0) * HH * WW;
            for (int r = 0; r < 30; ++r) {
                int y = y0 - PAD + r;
                float v = 0.f;
                if (zok && xok && (unsigned)y < (unsigned)HH)
                    v = src[y * WW + x];
                slabh[(s * 30 + r) * 32 + lcol] = (_Float16)v;
            }
        }
        __syncthreads();

        // ---- branch-free accumulation: 4 independent WMMA chains ----
        for (int dz = 0; dz < KS; ++dz) {
            for (int dy = 0; dy < KS; ++dy) {
                const v16h* bp = (const v16h*)(btab +
                    (((size_t)(dz * KS + dy) * 32 + lane) * 16));
                v16h bw = *bp;
                const int rbase = (dy + m) * 2 + half;
#pragma unroll
                for (int zo = 0; zo < ZT; ++zo) {
                    v16h a = slabv[(zo + dz) * 60 + rbase];
                    acc[t][zo] = __builtin_amdgcn_wmma_f32_16x16x32_f16(
                        false, a, false, bw, (short)0, acc[t][zo], false, false);
                }
            }
        }
        __syncthreads();   // before restaging LDS for the next tensor
    }

    // |conv(pred) - conv(target)|, fixed order -> deterministic
    float s = 0.f;
#pragma unroll
    for (int zo = 0; zo < ZT; ++zo)
#pragma unroll
        for (int i = 0; i < 8; ++i)
            s += fabsf(acc[0][zo][i] - acc[1][zo][i]);

#pragma unroll
    for (int off = 16; off > 0; off >>= 1)
        s += __shfl_down(s, off, 32);
    if (lane == 0) partial[blockIdx.x] = s;
}

// ---------------------------------------------------------------------------
// Phase 2: deterministic tree reduction of block partials -> mean
// ---------------------------------------------------------------------------
__global__ __launch_bounds__(256) void hfen_reduce(
    const float* __restrict__ partial, int n, float* __restrict__ out, float invN) {
    __shared__ float sh[256];
    float s = 0.f;
    for (int i = threadIdx.x; i < n; i += 256) s += partial[i];
    sh[threadIdx.x] = s;
    __syncthreads();
    for (int k = 128; k > 0; k >>= 1) {
        if ((int)threadIdx.x < k) sh[threadIdx.x] += sh[threadIdx.x + k];
        __syncthreads();
    }
    if (threadIdx.x == 0) out[0] = sh[0] * invN;
}

extern "C" void kernel_launch(void* const* d_in, const int* in_sizes, int n_in,
                              void* d_out, int out_size, void* d_ws, size_t ws_size,
                              hipStream_t stream) {
    const float* pred   = (const float*)d_in[0];
    const float* target = (const float*)d_in[1];

    _Float16* btab    = (_Float16*)d_ws;
    float*    partial = (float*)((char*)d_ws + BTAB_ELEMS * sizeof(_Float16));

    hfen_build_btable<<<1, 256, 0, stream>>>(btab);
    hfen_conv<<<NBLOCKS, 32, 0, stream>>>(pred, target, btab, partial);
    const float invN = 1.f / ((float)NB * DD * HH * WW);
    hfen_reduce<<<1, 256, 0, stream>>>(partial, NBLOCKS, (float*)d_out, invN);
}
